// ConvBNReLU1D_68968584839252
// MI455X (gfx1250) — matrix operand order repaired and checked by reading, not compiled, not run on hardware
//
#include <hip/hip_runtime.h>
#include <hip/hip_bf16.h>

// ---------------------------------------------------------------------------
// ConvBNReLU1D on MI455X (gfx1250): int8-fake-quant W is exact in bf16, so the
// GEMM runs on v_wmma_f32_16x16x32_bf16 with an x = hi+lo bf16 split for
// near-fp32 accuracy. Memory-bound target: ~256 MB @ 23.3 TB/s.
// A-tiles (quantized W) are DMA'd to LDS via global_load_async_to_lds_b128
// (ASYNCcnt path), else read direct from L2.
// ---------------------------------------------------------------------------

typedef __attribute__((ext_vector_type(16))) __bf16 v16bf;
typedef __attribute__((ext_vector_type(8)))  float  v8f;
typedef __attribute__((ext_vector_type(4)))  int    v4i;

union Frag {           // 32 bytes: one WMMA 16-bit A/B fragment per lane
    v16bf v;
    int4  i4[2];
};

__device__ __forceinline__ unsigned bf16_rne(float f) {
    unsigned u = __float_as_uint(f);
    return (u + 0x7FFFu + ((u >> 16) & 1u)) >> 16;   // round-to-nearest-even
}

#if defined(__gfx1250__) && __has_builtin(__builtin_amdgcn_global_load_async_to_lds_b128)
#define USE_ASYNC_LDS 1
#else
#define USE_ASYNC_LDS 0
#endif

#if USE_ASYNC_LDS
typedef __attribute__((address_space(1))) v4i gv4i;   // global 16B chunk
typedef __attribute__((address_space(3))) v4i lv4i;   // LDS 16B chunk

__device__ __forceinline__ void async_copy_b128(const unsigned short* g, int4* l) {
    __builtin_amdgcn_global_load_async_to_lds_b128(
        (gv4i*)g, (lv4i*)l, 0, 0);
}
#endif

// ---------------------------------------------------------------------------
// Kernel 1: max-abs stats -> power-of-two scales; fold bias+BN into per-channel
// alpha (scale) and bias2 (shift). One block of 256 threads.
// ---------------------------------------------------------------------------
__global__ __launch_bounds__(256) void prep_kernel(
    const float* __restrict__ W, const float* __restrict__ b,
    const float* __restrict__ gamma, const float* __restrict__ beta,
    const float* __restrict__ rmean, const float* __restrict__ rvar,
    float* __restrict__ alpha, float* __restrict__ bias2,
    float* __restrict__ swout)
{
    __shared__ float red[256];
    const int t = threadIdx.x;

    float m = 0.0f;
    for (int i = t; i < 256 * 256; i += 256) m = fmaxf(m, fabsf(W[i]));
    red[t] = m;
    __syncthreads();
    for (int s = 128; s > 0; s >>= 1) {
        if (t < s) red[t] = fmaxf(red[t], red[t + s]);
        __syncthreads();
    }
    const float maxW = red[0];
    __syncthreads();

    red[t] = fabsf(b[t]);
    __syncthreads();
    for (int s = 128; s > 0; s >>= 1) {
        if (t < s) red[t] = fmaxf(red[t], red[t + s]);
        __syncthreads();
    }
    const float maxB = red[0];

    const float s_w = exp2f(ceilf(log2f(fmaxf(maxW, 1e-12f) / 127.0f)));
    const float s_b = exp2f(ceilf(log2f(fmaxf(maxB, 1e-12f) / 127.0f)));

    const float bq  = fminf(fmaxf(rintf(b[t] / s_b), -128.0f), 127.0f) * s_b;
    const float inv = gamma[t] * rsqrtf(rvar[t] + 1e-5f);
    alpha[t] = inv;
    bias2[t] = (bq - rmean[t]) * inv + beta[t];
    if (t == 0) swout[0] = s_w;
}

// ---------------------------------------------------------------------------
// Kernel 2: fake-quant W (exactly representable in bf16: int8 value * 2^e).
// ---------------------------------------------------------------------------
__global__ __launch_bounds__(256) void quantW_kernel(
    const float* __restrict__ W, const float* __restrict__ sw,
    unsigned short* __restrict__ Wq)
{
    const int i = blockIdx.x * 256 + threadIdx.x;
    const float s = sw[0];
    const float q = fminf(fmaxf(rintf(W[i] / s), -128.0f), 127.0f) * s;
    Wq[i] = (unsigned short)bf16_rne(q);
}

// ---------------------------------------------------------------------------
// Kernel 3: WMMA GEMM. Block = 256 threads = 8 waves (2M x 4N), 128x128 tile.
// K stepped by 32, fully unrolled so the scheduler can pipeline async DMA,
// x staging, and WMMA across slabs. x staged to LDS as packed bf16 pairs,
// n-major with a 20-dword column stride so B fragments are two aligned
// ds_load_b128. A tile async-DMA'd to LDS (80 B row stride).
// ---------------------------------------------------------------------------
__global__ __launch_bounds__(256) void conv_wmma_kernel(
    const float* __restrict__ x, const unsigned short* __restrict__ Wq,
    const float* __restrict__ alpha, const float* __restrict__ bias2,
    const float* __restrict__ act, float* __restrict__ out)
{
    __shared__ int4 ldsB[2][128 * 5];   // [hi/lo][n * 5 int4 = 20 dwords]
#if USE_ASYNC_LDS
    __shared__ int4 ldsA[128 * 5];      // [m * 5 int4]: 32 bf16 data + pad
#endif

    const float act_s = act[0];
    const int tid   = threadIdx.x;
    const int col0  = blockIdx.x * 128;      // global column in [0, 32*4096)
    const int bi    = col0 >> 12;            // batch index
    const int n_in  = col0 & 4095;           // position within batch
    const int m0    = blockIdx.y * 128;      // output-channel block
    const int lane  = tid & 31;
    const int wid   = tid >> 5;
    const int half  = lane >> 4;             // lane group 0/1
    const int l16   = lane & 15;
    const int waveM = wid & 1;               // 2 M-strips of 64
    const int waveN = wid >> 1;              // 4 N-strips of 32

    const v8f zero8 = {0.f, 0.f, 0.f, 0.f, 0.f, 0.f, 0.f, 0.f};
    v8f acc[4][2];
    for (int mt = 0; mt < 4; ++mt)
        for (int nt = 0; nt < 2; ++nt) acc[mt][nt] = zero8;

    const float* xb = x + (size_t)bi * 256 * 4096;
    int* dst_hi = reinterpret_cast<int*>(&ldsB[0][0]);
    int* dst_lo = reinterpret_cast<int*>(&ldsB[1][0]);

#pragma unroll
    for (int k0 = 0; k0 < 256; k0 += 32) {
        __syncthreads();

#if USE_ASYNC_LDS
        // ---- async DMA the 128x32 bf16 A tile into LDS (512 b128 chunks) --
#pragma unroll
        for (int i = 0; i < 2; ++i) {
            const int id = i * 256 + tid;
            const int m  = id >> 2;            // row 0..127
            const int c  = id & 3;             // 16B chunk 0..3
            const unsigned short* gsrc = Wq + (size_t)(m0 + m) * 256 + k0 + c * 8;
            async_copy_b128(gsrc, &ldsA[m * 5 + c]);
        }
#endif

        // ---- stage x[k0:k0+32, n_in:n_in+128] as hi/lo bf16-pair dwords ----
#pragma unroll
        for (int i = 0; i < 8; ++i) {
            const int idx = i * 256 + tid;
            const int kp  = idx >> 7;          // k-pair 0..15
            const int n   = idx & 127;
            const float* px = xb + (size_t)(k0 + 2 * kp) * 4096 + (n_in + n);
            const float x0 = px[0];
            const float x1 = px[4096];
            if (k0 + 32 < 256)
                __builtin_prefetch(px + 32 * 4096, 0, 1);   // global_prefetch_b8
            const unsigned h0 = bf16_rne(x0), h1 = bf16_rne(x1);
            const float r0 = x0 - __uint_as_float(h0 << 16);
            const float r1 = x1 - __uint_as_float(h1 << 16);
            const unsigned l0 = bf16_rne(r0), l1 = bf16_rne(r1);
            dst_hi[n * 20 + kp] = (int)(h0 | (h1 << 16));
            dst_lo[n * 20 + kp] = (int)(l0 | (l1 << 16));
        }

#if USE_ASYNC_LDS
        asm volatile("s_wait_asynccnt 0" ::: "memory");
#endif
        __syncthreads();

        // ---- A fragments: contiguous layout -> K = 16*half .. 16*half+15
        Frag af[4];
#pragma unroll
        for (int mt = 0; mt < 4; ++mt) {
            const int m = waveM * 64 + mt * 16 + l16;
#if USE_ASYNC_LDS
            af[mt].i4[0] = ldsA[m * 5 + 2 * half];       // K = 16*half .. +7
            af[mt].i4[1] = ldsA[m * 5 + 2 * half + 1];   // K = 16*half+8 .. +7
#else
            const int4* gp = reinterpret_cast<const int4*>(
                Wq + (size_t)(m0 + m) * 256 + k0 + 16 * half);
            af[mt].i4[0] = gp[0];
            af[mt].i4[1] = gp[1];
#endif
        }

        // ---- B fragments + 16 WMMAs -------------------------------------
#pragma unroll
        for (int nt = 0; nt < 2; ++nt) {
            const int ncol = waveN * 32 + nt * 16 + l16;
            Frag bh, bl;
            const int4* ph = &ldsB[0][ncol * 5 + 2 * half];
            bh.i4[0] = ph[0]; bh.i4[1] = ph[1];   // K = 16*half .. 16*half+15
            const int4* pl = &ldsB[1][ncol * 5 + 2 * half];
            bl.i4[0] = pl[0]; bl.i4[1] = pl[1];
#pragma unroll
            for (int mt = 0; mt < 4; ++mt) {
                acc[mt][nt] = __builtin_amdgcn_wmma_f32_16x16x32_bf16(
                    false, af[mt].v, false, bh.v, (short)0, acc[mt][nt],
                    false, false);
                acc[mt][nt] = __builtin_amdgcn_wmma_f32_16x16x32_bf16(
                    false, af[mt].v, false, bl.v, (short)0, acc[mt][nt],
                    false, false);
            }
        }
    }

    // ---- epilogue: BN fold + ReLU + 8-bit unsigned fake-quant -------------
#pragma unroll
    for (int mt = 0; mt < 4; ++mt) {
#pragma unroll
        for (int nt = 0; nt < 2; ++nt) {
            const int n_g = n_in + waveN * 32 + nt * 16 + l16;
            float* po = out + (size_t)bi * 256 * 4096 + (size_t)n_g;
#pragma unroll
            for (int r = 0; r < 8; ++r) {
                const int o = m0 + waveM * 64 + mt * 16 + r + 8 * half;
                float v = acc[mt][nt][r] * alpha[o] + bias2[o];
                v = fmaxf(v, 0.0f);
                float q = rintf(v / act_s);
                q = fminf(fmaxf(q, 0.0f), 255.0f);
                po[(size_t)o * 4096] = q * act_s;
            }
        }
    }
}

// ---------------------------------------------------------------------------
extern "C" void kernel_launch(void* const* d_in, const int* in_sizes, int n_in,
                              void* d_out, int out_size, void* d_ws, size_t ws_size,
                              hipStream_t stream) {
    (void)in_sizes; (void)n_in; (void)out_size; (void)ws_size;
    const float* x     = (const float*)d_in[0];
    const float* W     = (const float*)d_in[1];
    const float* b     = (const float*)d_in[2];
    const float* gamma = (const float*)d_in[3];
    const float* beta  = (const float*)d_in[4];
    const float* rmean = (const float*)d_in[5];
    const float* rvar  = (const float*)d_in[6];
    const float* act   = (const float*)d_in[7];
    float* out = (float*)d_out;

    float* alpha = (float*)d_ws;                 // 256 f32
    float* bias2 = alpha + 256;                  // 256 f32
    float* sw    = alpha + 512;                  // 1 f32
    unsigned short* Wq = (unsigned short*)((char*)d_ws + 4096);  // 256*256 bf16

    prep_kernel<<<1, 256, 0, stream>>>(W, b, gamma, beta, rmean, rvar,
                                       alpha, bias2, sw);
    quantW_kernel<<<256, 256, 0, stream>>>(W, sw, Wq);
    conv_wmma_kernel<<<dim3(1024, 2, 1), 256, 0, stream>>>(x, Wq, alpha, bias2,
                                                           act, out);
}
